// MultiHeadattention_24361054502928
// MI455X (gfx1250) — matrix operand order repaired and checked by reading, not compiled, not run on hardware
//
#include <hip/hip_runtime.h>

// ---------------- problem constants ----------------
#define DM    1024
#define NH    16
#define DK    64
#define SEQ   2048
#define BATCH 2
#define MTOT  (BATCH * SEQ)   // 4096 flattened (b,s) rows

typedef __attribute__((ext_vector_type(16))) __bf16 v16bf;
typedef __attribute__((ext_vector_type(8)))  float  v8f;

union FragB { v16bf v; unsigned int u[8]; unsigned short s[16]; };
union FragC { v8f   v; float f[8]; };

__device__ __forceinline__ unsigned short f32_to_bf16_rne(float x) {
  union { float f; unsigned int u; } c; c.f = x;
  unsigned int r = c.u + 0x7fffu + ((c.u >> 16) & 1u);  // round-to-nearest-even
  return (unsigned short)(r >> 16);
}

// ---- CDNA5 async global->LDS copy (ASYNCcnt path, ISA §15.18.3 op 98) ----
// Per-lane: copies 16 bytes from global address g to LDS address lds.
__device__ __forceinline__ void async_ld_b128(unsigned lds, const void* g) {
  asm volatile("global_load_async_to_lds_b128 %0, %1, off"
               :: "v"(lds), "v"((unsigned long long)(uintptr_t)g) : "memory");
}
__device__ __forceinline__ void wait_async0() {
  asm volatile("s_wait_asynccnt 0x0" ::: "memory");
}
__device__ __forceinline__ unsigned lds_addr(const void* p) {
  return (unsigned)(uintptr_t)p;  // flat->LDS: addr[31:0] is the LDS offset
}

// ======================================================================
// One-pass f32 -> bf16 conversion (x and the four weight matrices).
// Each element converted exactly once; everything downstream moves b128.
// ======================================================================
__global__ __launch_bounds__(256)
void to_bf16(const float* __restrict__ in, unsigned short* __restrict__ out, int n4) {
  int i = blockIdx.x * 256 + threadIdx.x;
  if (i < n4) {
    float4 f = ((const float4*)in)[i];
    ushort4 o;
    o.x = f32_to_bf16_rne(f.x);
    o.y = f32_to_bf16_rne(f.y);
    o.z = f32_to_bf16_rne(f.z);
    o.w = f32_to_bf16_rne(f.w);
    ((ushort4*)out)[i] = o;
  }
}

// ======================================================================
// GEMM: Y[m,n] = sum_k A[m,k] * W[n,k] + bias[n]   (A, W already bf16)
// MODE 0: emit bf16 head-major [B][H][S][DK]  (QKV)   MODE 1: f32 [m][n]
// Block: 256 threads (8 waves). Tile: BM=256, BN=64, BK=32; wave owns two
// 16-row M-tiles x 4 N-tiles = 8 WMMAs per stage (B fragments reused 2x).
// DOUBLE-BUFFERED: next K-slice streams via global_load_async_to_lds_b128
// while WMMAs consume the current buffer; s_wait_asynccnt + one barrier.
// LDS stride 40 ushorts = 80 B: 16B-aligned for b128 AND conflict-free
// fragment reads (20*r mod 64 distinct for r = 0..15).
// ======================================================================
template <int MODE>
__global__ __launch_bounds__(256)
void proj_gemm(const unsigned short* __restrict__ Ab,   // [MTOT][DM] bf16
               const unsigned short* __restrict__ Wb,   // [DM][DM] bf16 ([n][k])
               const float* __restrict__ bias,          // [DM]
               unsigned short* __restrict__ outB,
               float* __restrict__ outF)
{
  __shared__ __attribute__((aligned(16))) unsigned short As[2][256][40]; // 40 KB
  __shared__ __attribute__((aligned(16))) unsigned short Bs[2][64][40];  // 10 KB

  const int tid  = threadIdx.x;
  const int w    = tid >> 5;
  const int lane = tid & 31;
  const int half = lane >> 4;
  const int l16  = lane & 15;
  const int m0   = blockIdx.x * 256;
  const int n0   = blockIdx.y * 64;

  // async-stage one 32-wide K slice into buffer `buf`
  auto stage = [&](int buf, int k0) {
#pragma unroll
    for (int i = 0; i < 4; ++i) {                    // A: 256x32 = 1024 b128
      int idx = i * 256 + tid;
      int r = idx >> 2, c = (idx & 3) * 8;
      async_ld_b128(lds_addr(&As[buf][r][c]),
                    &Ab[(size_t)(m0 + r) * DM + k0 + c]);
    }
    {                                                // B: 64x32 = 256 b128
      int r = tid >> 2, c = (tid & 3) * 8;
      async_ld_b128(lds_addr(&Bs[buf][r][c]),
                    &Wb[(size_t)(n0 + r) * DM + k0 + c]);
    }
  };

  FragC acc[2][4];
#pragma unroll
  for (int mt = 0; mt < 2; ++mt)
#pragma unroll
    for (int nt = 0; nt < 4; ++nt)
#pragma unroll
      for (int i = 0; i < 8; ++i) acc[mt][nt].f[i] = 0.f;

  stage(0, 0);
  wait_async0();
  __syncthreads();

  for (int k0 = 0; k0 < DM; k0 += 32) {
    const int cur = (k0 >> 5) & 1;
    if (k0 + 32 < DM) stage(cur ^ 1, k0 + 32);   // overlap next stage w/ math

    // ---- A fragments: contiguous K order K(e)=e+half*16, same as B ----
    FragB a[2];
#pragma unroll
    for (int mt = 0; mt < 2; ++mt) {
      const int ar = w * 32 + mt * 16 + l16;
#pragma unroll
      for (int j = 0; j < 8; ++j) {
        int e  = 2 * j;
        int kk = e + 16 * half;
        a[mt].u[j] = *(const unsigned int*)&As[cur][ar][kk];
      }
    }

#pragma unroll
    for (int nt = 0; nt < 4; ++nt) {
      FragB bfr;                        // B 32x16: K = e + 16*half, N = l16
      const int br = nt * 16 + l16;
#pragma unroll
      for (int j = 0; j < 8; ++j) {
        int kk = 2 * j + 16 * half;
        bfr.u[j] = *(const unsigned int*)&Bs[cur][br][kk];
      }
      acc[0][nt].v = __builtin_amdgcn_wmma_f32_16x16x32_bf16(
          false, a[0].v, false, bfr.v, (short)0, acc[0][nt].v, false, false);
      acc[1][nt].v = __builtin_amdgcn_wmma_f32_16x16x32_bf16(
          false, a[1].v, false, bfr.v, (short)0, acc[1][nt].v, false, false);
    }

    wait_async0();       // next buffer fully landed (wave's own async ops)
    __syncthreads();     // all waves done staging + done reading `cur`
  }

  // ---- epilogue: C layout M = r + 8*half, N = l16 ----
#pragma unroll
  for (int mt = 0; mt < 2; ++mt)
#pragma unroll
    for (int nt = 0; nt < 4; ++nt) {
      int n = n0 + nt * 16 + l16;
      float bv = bias[n];
#pragma unroll
      for (int r = 0; r < 8; ++r) {
        int m = m0 + w * 32 + mt * 16 + r + 8 * half;
        float v = acc[mt][nt].f[r] + bv;
        if (MODE == 0) {
          int b = m >> 11, s = m & (SEQ - 1);
          int h = n >> 6,  dk = n & (DK - 1);
          outB[(((size_t)b * NH + h) * SEQ + s) * DK + dk] = f32_to_bf16_rne(v);
        } else {
          outF[(size_t)m * DM + n] = v;
        }
      }
    }
}

// ======================================================================
// Fused attention, softmax over the HEAD axis (reference quirk):
//   attn[b,h,q,k] = exp(s[b,h,q,k]) / sum_h' exp(s[b,h',q,k])
// Reduction is per-(q,k) across heads -> tile-local; scores never touch
// global memory (saves ~1 GB HBM traffic, ~44 us at 23.3 TB/s).
// Block: 512 threads = 16 waves; wave h owns head h for one 16-row q-tile.
// V tiles stream via async-to-LDS overlapping the whole score/exp phase;
// denominator via ds_add_f32 atomics, inverted ONCE per element (v_rcp).
// ======================================================================
__global__ __launch_bounds__(512)
void attn_kernel(const unsigned short* __restrict__ Qb,   // [B][H][S][DK] bf16
                 const unsigned short* __restrict__ Kb,
                 const unsigned short* __restrict__ Vb,
                 unsigned short* __restrict__ ctxb)        // [MTOT][DM] bf16
{
  __shared__ float sden[512];                                              // 2 KB
  __shared__ __attribute__((aligned(16))) unsigned short abuf[NH][16][32]; // 16 KB
  __shared__ __attribute__((aligned(16))) unsigned short vbuf[NH][32][64]; // 64 KB

  const int tid  = threadIdx.x;
  const int h    = tid >> 5;            // wave id == head id
  const int lane = tid & 31;
  const int half = lane >> 4;
  const int l16  = lane & 15;
  const int q0   = blockIdx.x * 16;
  const int b    = blockIdx.y;
  const size_t headBase = ((size_t)b * NH + h) * SEQ * DK;

  // ---- Q A-fragments for this head's q-tile: 16x64 = two K=32 chunks ----
  FragB qa[2];
  {
    const unsigned short* qrow = Qb + headBase + (size_t)(q0 + l16) * DK;
#pragma unroll
    for (int c = 0; c < 2; ++c)
#pragma unroll
      for (int j = 0; j < 8; ++j) {
        int e = 2 * j;
        int d = c * 32 + e + 16 * half;
        qa[c].u[j] = *(const unsigned int*)(qrow + d);
      }
  }

  FragC cacc[4];                         // context 16q x 64d accumulators
#pragma unroll
  for (int dt = 0; dt < 4; ++dt)
#pragma unroll
    for (int i = 0; i < 8; ++i) cacc[dt].f[i] = 0.f;

  const float scale = 0.125f;            // 1/sqrt(DK)

  for (int k0 = 0; k0 < SEQ; k0 += 32) {
    __syncthreads();                     // protect sden reuse
    sden[tid] = 0.f;
    __syncthreads();

    // ---- kick off async V-tile staging: overlaps entire score phase ----
    // (prev iteration's vbuf reads completed before the loop-top barrier)
    {
      unsigned lbase = lds_addr(&vbuf[h][lane][0]);
      const unsigned short* g = Vb + headBase + (size_t)(k0 + lane) * DK;
#pragma unroll
      for (int i = 0; i < 8; ++i)
        async_ld_b128(lbase + i * 16, g + i * 8);
    }
    // prefetch next k-step's K rows (one 128B row per lane)
    if (k0 + 32 < SEQ)
      __builtin_prefetch(Kb + headBase + (size_t)(k0 + 32 + lane) * DK, 0, 0);

    // ---- load ALL K B-fragments first (one clause), then 4 WMMAs ----
    FragB kb[2][2];  // [kk-tile][d-chunk]; B: K-dim = d, N = kk
#pragma unroll
    for (int t = 0; t < 2; ++t) {
      const unsigned short* krow = Kb + headBase + (size_t)(k0 + t * 16 + l16) * DK;
#pragma unroll
      for (int j = 0; j < 8; ++j) {
        int d = 2 * j + 16 * half;
        kb[t][0].u[j] = *(const unsigned int*)(krow + d);
        kb[t][1].u[j] = *(const unsigned int*)(krow + 32 + d);
      }
    }
    FragC s[2];
#pragma unroll
    for (int t = 0; t < 2; ++t) {
#pragma unroll
      for (int i = 0; i < 8; ++i) s[t].f[i] = 0.f;
      s[t].v = __builtin_amdgcn_wmma_f32_16x16x32_bf16(
          false, qa[0].v, false, kb[t][0].v, (short)0, s[t].v, false, false);
      s[t].v = __builtin_amdgcn_wmma_f32_16x16x32_bf16(
          false, qa[1].v, false, kb[t][1].v, (short)0, s[t].v, false, false);
    }

    // exp (no max-subtraction: head-axis softmax is identical without it;
    // scores are O(1)) + cross-head denominator via ds_add_f32
    float p[2][8];
#pragma unroll
    for (int t = 0; t < 2; ++t)
#pragma unroll
      for (int r = 0; r < 8; ++r) {
        float pe = __expf(s[t].f[r] * scale);
        p[t][r]  = pe;
        int eid  = (t * 16 + l16) * 16 + (r + 8 * half);  // kk*16 + m
        atomicAdd(&sden[eid], pe);
      }

    wait_async0();                       // V tile landed in vbuf[h]
    __syncthreads();                     // denominator complete across heads
    sden[tid] = __builtin_amdgcn_rcpf(sden[tid]);  // invert once per element
    __syncthreads();

    // ---- attn = p * rinv, restage as bf16 A-matrix (16x32) in LDS ----
#pragma unroll
    for (int t = 0; t < 2; ++t)
#pragma unroll
      for (int r = 0; r < 8; ++r) {
        int kk = t * 16 + l16;
        int m  = r + 8 * half;
        float av = p[t][r] * sden[kk * 16 + m];
        abuf[h][m][kk] = f32_to_bf16_rne(av);
      }

    FragB aa;                            // wave-local RAW: compiler waits DScnt
#pragma unroll
    for (int j = 0; j < 8; ++j) {
      int e  = 2 * j;
      int kk = e + 16 * half;
      aa.u[j] = *(const unsigned int*)&abuf[h][l16][kk];
    }

    // ---- context += attn(16x32) @ V(32x64): 4 d-tiles ----
#pragma unroll
    for (int dt = 0; dt < 4; ++dt) {
      FragB vb;                          // B: K = kk = e + 16*half, N = dd
#pragma unroll
      for (int e = 0; e < 16; ++e) {
        int kk = e + 16 * half;
        vb.s[e] = vbuf[h][kk][dt * 16 + l16];
      }
      cacc[dt].v = __builtin_amdgcn_wmma_f32_16x16x32_bf16(
          false, aa.v, false, vb.v, (short)0, cacc[dt].v, false, false);
    }
  }

  // ---- write context in [b,s, h*64+d] bf16 layout (feeds O-proj GEMM) ----
#pragma unroll
  for (int dt = 0; dt < 4; ++dt) {
    int d = h * DK + dt * 16 + l16;
#pragma unroll
    for (int r = 0; r < 8; ++r) {
      int m = b * SEQ + q0 + r + 8 * half;
      ctxb[(size_t)m * DM + d] = f32_to_bf16_rne(cacc[dt].f[r]);
    }
  }
}

// ======================================================================
extern "C" void kernel_launch(void* const* d_in, const int* in_sizes, int n_in,
                              void* d_out, int out_size, void* d_ws, size_t ws_size,
                              hipStream_t stream) {
  (void)in_sizes; (void)n_in; (void)out_size; (void)ws_size;

  const float* x  = (const float*)d_in[0];
  const float* Wq = (const float*)d_in[1];
  const float* bq = (const float*)d_in[2];
  const float* Wk = (const float*)d_in[3];
  const float* bk = (const float*)d_in[4];
  const float* Wv = (const float*)d_in[5];
  const float* bv = (const float*)d_in[6];
  const float* Wo = (const float*)d_in[7];
  const float* bo = (const float*)d_in[8];
  float* out = (float*)d_out;

  // ws layout (ushort units): xb 4M | Wqb/Wkb/Wvb/Wob 1M each | Qb/Kb/Vb 4M | ctxb 4M
  unsigned short* ws = (unsigned short*)d_ws;
  unsigned short* xb   = ws;                  // 8 MiB
  unsigned short* Wqb  = xb   + (size_t)MTOT * DM;
  unsigned short* Wkb  = Wqb  + (size_t)DM * DM;
  unsigned short* Wvb  = Wkb  + (size_t)DM * DM;
  unsigned short* Wob  = Wvb  + (size_t)DM * DM;
  unsigned short* Qb   = Wob  + (size_t)DM * DM;
  unsigned short* Kb   = Qb   + (size_t)BATCH * NH * SEQ * DK;
  unsigned short* Vb   = Kb   + (size_t)BATCH * NH * SEQ * DK;
  unsigned short* ctxb = Vb   + (size_t)BATCH * NH * SEQ * DK;  // total 48 MiB

  const int n4x = MTOT * DM / 4, n4w = DM * DM / 4;
  to_bf16<<<(n4x + 255) / 256, 256, 0, stream>>>(x,  xb,  n4x);
  to_bf16<<<(n4w + 255) / 256, 256, 0, stream>>>(Wq, Wqb, n4w);
  to_bf16<<<(n4w + 255) / 256, 256, 0, stream>>>(Wk, Wkb, n4w);
  to_bf16<<<(n4w + 255) / 256, 256, 0, stream>>>(Wv, Wvb, n4w);
  to_bf16<<<(n4w + 255) / 256, 256, 0, stream>>>(Wo, Wob, n4w);

  dim3 g(MTOT / 256, DM / 64);  // 16 x 16 blocks
  proj_gemm<0><<<g, 256, 0, stream>>>(xb, Wqb, bq, Qb, nullptr);
  proj_gemm<0><<<g, 256, 0, stream>>>(xb, Wkb, bk, Kb, nullptr);
  proj_gemm<0><<<g, 256, 0, stream>>>(xb, Wvb, bv, Vb, nullptr);

  attn_kernel<<<dim3(SEQ / 16, BATCH), 512, 0, stream>>>(Qb, Kb, Vb, ctxb);

  proj_gemm<1><<<g, 256, 0, stream>>>(ctxb, Wob, bo, nullptr, out);
}
